// stackedIndRNN_encoder_7318624272387
// MI455X (gfx1250) — matrix operand order repaired and checked by reading, not compiled, not run on hardware
//
#include <hip/hip_runtime.h>

// ---------------------------------------------------------------------------
// Problem dims (from reference)
// ---------------------------------------------------------------------------
#define TT 512
#define BB 128
#define JJ 25
#define HH 512
#define LL 6
#define CC 60
#define K0PAD 96            // layer-0 K (3*J=75) padded to multiple of 32
#define EPS 1e-5f

typedef __attribute__((ext_vector_type(16))) __bf16 v16bf;
typedef __attribute__((ext_vector_type(8)))  float  v8f;

__device__ __forceinline__ unsigned short f32_to_bf16(float f) {
  unsigned int u = __float_as_uint(f);
  u += 0x7FFFu + ((u >> 16) & 1u);      // round-to-nearest-even
  return (unsigned short)(u >> 16);
}
__device__ __forceinline__ float bf16_to_f32(unsigned short h) {
  return __uint_as_float(((unsigned int)h) << 16);
}

// LDS byte offset (wave-relative) of a __shared__ object: generic -> AS(3)
typedef __attribute__((address_space(3))) unsigned char lds_byte_t;
__device__ __forceinline__ unsigned lds_off(void* p) {
  return (unsigned)(unsigned long long)(lds_byte_t*)p;
}

// Async global -> LDS copy, 16B per lane, tracked by ASYNCcnt.
__device__ __forceinline__ void async_g2l_b128(unsigned ldsoff, const void* gptr) {
  asm volatile("global_load_async_to_lds_b128 %0, %1, off"
               :: "v"(ldsoff), "v"((unsigned long long)gptr)
               : "memory");
}

// ---------------------------------------------------------------------------
// Pack kernels: fp32 -> bf16 (with K padding for layer 0)
// ---------------------------------------------------------------------------
__global__ void pack_pad_bf16(const float* __restrict__ src, unsigned short* __restrict__ dst,
                              int rows, int kin, int kout) {
  int i = blockIdx.x * blockDim.x + threadIdx.x;
  int n = rows * kout;
  if (i >= n) return;
  int r = i / kout, c = i - r * kout;
  float v = (c < kin) ? src[(size_t)r * kin + c] : 0.0f;
  dst[i] = f32_to_bf16(v);
}

__global__ void pack_bf16(const float* __restrict__ src, unsigned short* __restrict__ dst, int n) {
  int i = blockIdx.x * blockDim.x + threadIdx.x;
  if (i < n) dst[i] = f32_to_bf16(src[i]);
}

// ---------------------------------------------------------------------------
// WMMA GEMM:  C[M,N] = A[M,K] * W[N,K]^T + bias[N]
//   A, W bf16 row-major; C fp32.
//   Block: 128x128 tile, 256 threads (8 waves); wave = 32(M) x 64(N).
//   Double-buffered LDS, async global->LDS staging (ASYNCcnt-tracked).
// ---------------------------------------------------------------------------
#define GTM 128
#define GTN 128
#define GTK 32

union AFrag { uint4 q[2]; v16bf v; };

__global__ __launch_bounds__(256) void gemm_bias_wmma(
    const unsigned short* __restrict__ A,   // [M x K]
    const unsigned short* __restrict__ W,   // [N x K]
    const float* __restrict__ bias,         // [N]
    float* __restrict__ C,                  // [M x N]
    int M, int N, int K) {
  __shared__ alignas(32) unsigned short sA[2][GTM * GTK];  // 2 x 8 KB
  __shared__ alignas(32) unsigned short sB[2][GTN * GTK];  // 2 x 8 KB

  int tid   = threadIdx.x;
  int lane  = tid & 31;
  int wave  = tid >> 5;       // 0..7
  int waveM = wave & 3;       // 4 waves along M (4*32 = 128)
  int waveN = wave >> 2;      // 2 waves along N (2*64 = 128)
  int blockM = blockIdx.x * GTM;
  int blockN = blockIdx.y * GTN;

  // Per-thread staging slots: 512 x 16B per tile, 2 slots per thread per array.
  int s0 = tid, s1 = tid + 256;
  int r0 = s0 >> 2, c0 = s0 & 3;
  int r1 = s1 >> 2, c1 = s1 & 3;

  auto stage = [&](int k0, int buf) {
    unsigned la = lds_off(&sA[buf][0]);
    unsigned lb = lds_off(&sB[buf][0]);
    async_g2l_b128(la + s0 * 16, A + (size_t)(blockM + r0) * K + k0 + c0 * 8);
    async_g2l_b128(lb + s0 * 16, W + (size_t)(blockN + r0) * K + k0 + c0 * 8);
    async_g2l_b128(la + s1 * 16, A + (size_t)(blockM + r1) * K + k0 + c1 * 8);
    async_g2l_b128(lb + s1 * 16, W + (size_t)(blockN + r1) * K + k0 + c1 * 8);
  };

  v8f c[2][4];
#pragma unroll
  for (int mt = 0; mt < 2; ++mt)
#pragma unroll
    for (int nt = 0; nt < 4; ++nt)
      c[mt][nt] = (v8f){0.f, 0.f, 0.f, 0.f, 0.f, 0.f, 0.f, 0.f};

  int g  = lane >> 4;   // half-wave selector
  int rr = lane & 15;   // row/col within 16

  stage(0, 0);          // prologue: tile 0 -> buffer 0 (4 async ops in flight)

  for (int i = 0, k0 = 0; k0 < K; ++i, k0 += GTK) {
    int buf = i & 1;
    bool hasNext = (k0 + GTK) < K;
    if (hasNext) {
      stage(k0 + GTK, buf ^ 1);   // overlap next tile's copy with this compute
      // ASYNCcnt completes in order: <=4 left means current buffer has landed.
      asm volatile("s_wait_asynccnt 0x4" ::: "memory");
    } else {
      asm volatile("s_wait_asynccnt 0x0" ::: "memory");
    }
    __syncthreads();              // cross-wave: everyone's writes to buf are visible

    // B fragments: 32x16 bf16, lane half g holds K = g*16 .. g*16+15 (contiguous 32B).
    v16bf bfr[4];
#pragma unroll
    for (int nt = 0; nt < 4; ++nt) {
      const unsigned short* p = &sB[buf][(size_t)(waveN * 64 + nt * 16 + rr) * GTK + g * 16];
      bfr[nt] = *(const v16bf*)p;
    }
    // A fragments: 16x32 bf16, lane half g holds K = g*16 .. g*16+15 (contiguous).
    AFrag afr[2];
#pragma unroll
    for (int mt = 0; mt < 2; ++mt) {
      const unsigned short* base = &sA[buf][(size_t)(waveM * 32 + mt * 16 + rr) * GTK];
      afr[mt].q[0] = *(const uint4*)(base + g * 16);
      afr[mt].q[1] = *(const uint4*)(base + g * 16 + 8);
    }
#pragma unroll
    for (int mt = 0; mt < 2; ++mt)
#pragma unroll
      for (int nt = 0; nt < 4; ++nt)
        c[mt][nt] = __builtin_amdgcn_wmma_f32_16x16x32_bf16(
            false, afr[mt].v, false, bfr[nt], (short)0, c[mt][nt], false, false);

    __syncthreads();              // all reads of buf done before it is re-staged
  }

  // Epilogue: C/D layout — element e of v8f: row = tile + g*8 + e, col = tile + (lane&15).
#pragma unroll
  for (int mt = 0; mt < 2; ++mt) {
#pragma unroll
    for (int nt = 0; nt < 4; ++nt) {
      int col = blockN + waveN * 64 + nt * 16 + rr;
      float bv = bias[col];
      int rowbase = blockM + waveM * 32 + mt * 16 + g * 8;
#pragma unroll
      for (int e = 0; e < 8; ++e)
        C[(size_t)(rowbase + e) * N + col] = c[mt][nt][e] + bv;
    }
  }
}

// ---------------------------------------------------------------------------
// IndRNN scan over T: h_t = relu(pre_t + u*h_{t-1}); one thread per (b,h).
// Writes bf16 activations; accumulates per-channel sum / sumsq (fp32 atomics).
// ---------------------------------------------------------------------------
__global__ __launch_bounds__(256) void indrnn_scan(
    const float* __restrict__ pre,          // [T, B*H]
    const float* __restrict__ u,            // [H]
    unsigned short* __restrict__ hbf,       // [T, B*H] bf16 out
    float* __restrict__ stats) {            // [2*H]: sum, sumsq
  int idx = blockIdx.x * blockDim.x + threadIdx.x;   // 0 .. B*H-1
  int hh  = idx & (HH - 1);
  float uv = u[hh];
  float h = 0.f, s = 0.f, s2 = 0.f;
  const int stride = BB * HH;
#pragma unroll 4
  for (int t = 0; t < TT; ++t) {
    float p = pre[(size_t)t * stride + idx];
    h = fmaxf(0.f, fmaf(uv, h, p));
    hbf[(size_t)t * stride + idx] = f32_to_bf16(h);
    s += h;
    s2 += h * h;
  }
  atomicAdd(&stats[hh], s);
  atomicAdd(&stats[HH + hh], s2);
}

// ---------------------------------------------------------------------------
// BatchNorm apply (in place on bf16 buffer)
// ---------------------------------------------------------------------------
__global__ __launch_bounds__(256) void bn_apply(
    unsigned short* __restrict__ hbf,       // [T*B*H]
    const float* __restrict__ stats,        // [2*H]
    const float* __restrict__ gamma,        // [H]
    const float* __restrict__ beta) {       // [H]
  size_t i = (size_t)blockIdx.x * blockDim.x + threadIdx.x;
  int hh = (int)(i & (HH - 1));
  const float inv_n = 1.0f / (float)(TT * BB);
  float mean = stats[hh] * inv_n;
  float var  = stats[HH + hh] * inv_n - mean * mean;
  float scl  = gamma[hh] * rsqrtf(var + EPS);
  float x = bf16_to_f32(hbf[i]);
  hbf[i] = f32_to_bf16((x - mean) * scl + beta[hh]);
}

// ---------------------------------------------------------------------------
// Final projection: out[B,C] = h_last[B,H] @ Wout[C,H]^T + bout
// ---------------------------------------------------------------------------
__global__ __launch_bounds__(256) void final_proj(
    const unsigned short* __restrict__ hlast,  // [B*H] bf16
    const float* __restrict__ Wout,            // [C,H]
    const float* __restrict__ bout,            // [C]
    float* __restrict__ out) {                 // [B,C]
  int i = blockIdx.x * blockDim.x + threadIdx.x;
  if (i >= BB * CC) return;
  int b = i / CC, c = i - b * CC;
  float acc = bout[c];
  const unsigned short* hp = hlast + (size_t)b * HH;
  const float* wp = Wout + (size_t)c * HH;
#pragma unroll 8
  for (int h = 0; h < HH; ++h) acc = fmaf(bf16_to_f32(hp[h]), wp[h], acc);
  out[i] = acc;
}

// ---------------------------------------------------------------------------
// Host-side orchestration
// ---------------------------------------------------------------------------
extern "C" void kernel_launch(void* const* d_in, const int* in_sizes, int n_in,
                              void* d_out, int out_size, void* d_ws, size_t ws_size,
                              hipStream_t stream) {
  const float* x     = (const float*)d_in[0];   // [T,B,J,3]
  const float* W0    = (const float*)d_in[1];   // [H, 75]
  const float* b0    = (const float*)d_in[2];   // [H]
  const float* Wh    = (const float*)d_in[3];   // [L-1,H,H]
  const float* bh    = (const float*)d_in[4];   // [L-1,H]
  const float* u     = (const float*)d_in[5];   // [L,H]
  const float* gamma = (const float*)d_in[6];   // [L,H]
  const float* beta  = (const float*)d_in[7];   // [L,H]
  const float* Wout  = (const float*)d_in[8];   // [C,H]
  const float* bout  = (const float*)d_in[9];   // [C]
  float* out = (float*)d_out;

  const size_t M = (size_t)TT * BB;             // 65536 rows
  char* ws = (char*)d_ws;
  size_t offHbf   = 0;                                    // M*H bf16      = 64 MB
  size_t offPre   = offHbf + M * HH * sizeof(unsigned short);
  size_t offW0b   = offPre + M * HH * sizeof(float);      // pre fp32      = 134 MB
  size_t offWhb   = offW0b + (size_t)HH * K0PAD * sizeof(unsigned short);
  size_t offStats = offWhb + (size_t)(LL - 1) * HH * HH * sizeof(unsigned short);

  unsigned short* hbf  = (unsigned short*)(ws + offHbf);
  float*          pre  = (float*)(ws + offPre);
  unsigned short* w0b  = (unsigned short*)(ws + offW0b);
  unsigned short* whb  = (unsigned short*)(ws + offWhb);
  float*          stats = (float*)(ws + offStats);

  // --- pack inputs / weights to bf16 ---
  {
    int n = (int)(M * K0PAD);
    pack_pad_bf16<<<(n + 255) / 256, 256, 0, stream>>>(x, hbf, (int)M, 3 * JJ, K0PAD);
  }
  {
    int n = HH * K0PAD;
    pack_pad_bf16<<<(n + 255) / 256, 256, 0, stream>>>(W0, w0b, HH, 3 * JJ, K0PAD);
  }
  {
    int n = (LL - 1) * HH * HH;
    pack_bf16<<<(n + 255) / 256, 256, 0, stream>>>(Wh, whb, n);
  }

  // --- stacked layers ---
  for (int l = 0; l < LL; ++l) {
    int K = (l == 0) ? K0PAD : HH;
    const unsigned short* Wl = (l == 0) ? w0b : (whb + (size_t)(l - 1) * HH * HH);
    const float* bl = (l == 0) ? b0 : (bh + (size_t)(l - 1) * HH);

    dim3 grid((unsigned)(M / GTM), (unsigned)(HH / GTN));
    gemm_bias_wmma<<<grid, 256, 0, stream>>>(hbf, Wl, bl, pre, (int)M, HH, K);

    hipMemsetAsync(stats, 0, 2 * HH * sizeof(float), stream);
    indrnn_scan<<<(BB * HH) / 256, 256, 0, stream>>>(pre, u + (size_t)l * HH, hbf, stats);

    size_t total = M * HH;
    bn_apply<<<(unsigned)(total / 256), 256, 0, stream>>>(
        hbf, stats, gamma + (size_t)l * HH, beta + (size_t)l * HH);
  }

  // --- final projection from last timestep ---
  const unsigned short* hlast = hbf + (size_t)(TT - 1) * BB * HH;
  final_proj<<<(BB * CC + 255) / 256, 256, 0, stream>>>(hlast, Wout, bout, out);
}
